// FusedDenseSqreluDense_22694607192381
// MI455X (gfx1250) — matrix operand order repaired and checked by reading, not compiled, not run on hardware
//
#include <hip/hip_runtime.h>
#include <hip/hip_bf16.h>

// ---------------------------------------------------------------------------
// Fused MLP: out = sqrelu(x @ w1^T + b1) @ w2^T + b2
//   x  : [M=8192, D=2048] fp32, w1 : [8192, 2048] fp32, w2 : [2048, 8192] fp32
// Compute-bound (~550 GFLOP vs ~0.4 GB min traffic): bf16 WMMA, f32 accum.
// CDNA5 path: global_load_async_to_lds_b128 double-buffered tiles (ASYNCcnt),
//             ds_load_b128 (offset-immediate) B fragments (no transpose),
//             v_wmma_f32_16x16x32_bf16 inner loop, 16 WMMA : 16 DS per K-step.
// ---------------------------------------------------------------------------

typedef __attribute__((ext_vector_type(4)))  float   f32x4;
typedef __attribute__((ext_vector_type(8)))  float   v8f;
typedef __attribute__((ext_vector_type(8)))  __bf16  bf16x8;
typedef __attribute__((ext_vector_type(16))) __bf16  bf16x16;

#define M_ROWS 8192   // B*S
#define D_DIM  2048
#define H_DIM  8192

// 256 threads = 8 waves (wave32), wave grid 2x4, wave tile 64x64.
#define BM 128
#define BN 256
#define BK 64
#define LDT (BK + 8)         // LDS row stride in halves (72 -> 144 B)
#define NT_STRIDE (16 * LDT * 2)   // bytes between 16-row N groups = 2304

// ---------------------------------------------------------------------------
// fp32 -> bf16 conversion
// ---------------------------------------------------------------------------
__global__ __launch_bounds__(256) void cvt_f32_to_bf16(
    const float* __restrict__ in, __bf16* __restrict__ out, int n4) {
  int i = blockIdx.x * blockDim.x + threadIdx.x;
  if (i < n4) {
    f32x4 v = *(const f32x4*)(in + (size_t)i * 4);
    __bf16* o = out + (size_t)i * 4;
    o[0] = (__bf16)v.x;
    o[1] = (__bf16)v.y;
    o[2] = (__bf16)v.z;
    o[3] = (__bf16)v.w;
  }
}

// ---------------------------------------------------------------------------
// Async global -> LDS staging of one ROWS x 64 bf16 tile (256 threads,
// ROWS/32 x b128 transfers per thread). Tracked by ASYNCcnt.
// ---------------------------------------------------------------------------
template <int ROWS>
__device__ __forceinline__ void issue_tile_async(
    const __bf16* __restrict__ G, int row0, int K, int k0,
    __bf16* lds_tile, int tid) {
#pragma unroll
  for (int i = 0; i < ROWS / 32; ++i) {
    int c   = tid + i * 256;      // chunk id
    int row = c >> 3;             // 0..ROWS-1
    int col = (c & 7) * 8;        // 0..56 (halves)
    const __bf16* g = G + (size_t)(row0 + row) * K + k0 + col;
    unsigned l = (unsigned)(size_t)(lds_tile + row * LDT + col);
    asm volatile("global_load_async_to_lds_b128 %0, %1, off"
                 :: "v"(l), "v"(g) : "memory");
  }
}

__device__ __forceinline__ bf16x16 concat8(bf16x8 lo, bf16x8 hi) {
  return __builtin_shufflevector(lo, hi, 0, 1, 2, 3, 4, 5, 6, 7,
                                 8, 9, 10, 11, 12, 13, 14, 15);
}

// Plain LDS b128 load with compile-time DS offset immediate (no per-load VALU).
template <int OFF>
__device__ __forceinline__ bf16x8 tr16(unsigned base) {
  bf16x8 d;
  asm volatile("ds_load_b128 %0, %1 offset:%c2"
               : "=v"(d) : "v"(base), "n"(OFF) : "memory");
  return d;
}

// ---------------------------------------------------------------------------
// WMMA bf16 GEMM:  Out[M,N] = A[M,K] * Bmat[N,K]^T + bias[N]
// EPILOGUE 0: Out = bf16( (relu(acc+bias))^2 )   (fc1 + squared ReLU)
// EPILOGUE 1: Out = fp32( acc + bias )           (fc2)
// ---------------------------------------------------------------------------
template <int EPILOGUE>
__global__ __launch_bounds__(256) void gemm_wmma_bf16(
    const __bf16* __restrict__ A,      // [M,K] row-major
    const __bf16* __restrict__ Bmat,   // [N,K] row-major
    const float*  __restrict__ bias,   // [N]
    void* __restrict__ Out,            // bf16 [M,N] (EPI 0) or f32 [M,N] (EPI 1)
    int M, int N, int K) {
  __shared__ __align__(16) __bf16 As[2][BM * LDT];   // 2 x 18 KB
  __shared__ __align__(16) __bf16 Bs[2][BN * LDT];   // 2 x 36 KB

  const int tid   = threadIdx.x;
  const int lane  = tid & 31;
  const int wid   = tid >> 5;        // 0..7
  const int waveM = wid >> 2;        // 0..1  -> 64-row slab
  const int waveN = wid & 3;         // 0..3  -> 64-col slab
  const int bm    = blockIdx.y * BM;
  const int bn    = blockIdx.x * BN;

  v8f acc[4][4] = {};

  // Prologue: stage first K-tile.
  issue_tile_async<BM>(A,    bm, K, 0, As[0], tid);
  issue_tile_async<BN>(Bmat, bn, K, 0, Bs[0], tid);
  asm volatile("s_wait_asynccnt 0x0" ::: "memory");
  __syncthreads();

  int buf = 0;
  for (int k0 = 0; k0 < K; k0 += BK) {
    // Prefetch next K-tile asynchronously into the other buffer.
    if (k0 + BK < K) {
      issue_tile_async<BM>(A,    bm, K, k0 + BK, As[buf ^ 1], tid);
      issue_tile_async<BN>(Bmat, bn, K, k0 + BK, Bs[buf ^ 1], tid);
    }

    // Per-wave base address for B fragment loads (lane&15 = N row within a
    // 16-row group, lanes 16..31 take the +8 K segment).
    unsigned bbase = (unsigned)(size_t)(
        &Bs[buf][(waveN * 64 + (lane & 15)) * LDT + ((lane >> 4) << 3)]);

#pragma unroll
    for (int ks = 0; ks < BK; ks += 32) {
      // A fragments (16-bit A 16x32 layout): lane&15 = M row, lanes 16..31
      // take K base +8; halves {K[b..b+7], K[b+16..b+23]}.
      bf16x16 afrag[4];
      const int mrow = lane & 15;
      const int kb   = ks + ((lane >> 4) << 3);
#pragma unroll
      for (int mt = 0; mt < 4; ++mt) {
        const __bf16* p = &As[buf][(waveM * 64 + mt * 16 + mrow) * LDT + kb];
        afrag[mt] = concat8(*(const bf16x8*)p, *(const bf16x8*)(p + 16));
      }
      // B fragments: per 32(K)x16(N) operand, two plain b128 loads (k 0..15, 16..31).
      // All offsets are DS-offset immediates off one base register.
      unsigned bk = bbase + (unsigned)(ks * 2);   // +64 B for second K-step
      bf16x16 bfrag[4];
      bfrag[0] = concat8(tr16<0 * NT_STRIDE +  0>(bk), tr16<0 * NT_STRIDE + 32>(bk));
      bfrag[1] = concat8(tr16<1 * NT_STRIDE +  0>(bk), tr16<1 * NT_STRIDE + 32>(bk));
      bfrag[2] = concat8(tr16<2 * NT_STRIDE +  0>(bk), tr16<2 * NT_STRIDE + 32>(bk));
      bfrag[3] = concat8(tr16<3 * NT_STRIDE +  0>(bk), tr16<3 * NT_STRIDE + 32>(bk));
      asm volatile("s_wait_dscnt 0x0" ::: "memory");
#pragma unroll
      for (int mt = 0; mt < 4; ++mt)
#pragma unroll
        for (int nt = 0; nt < 4; ++nt)
          acc[mt][nt] = __builtin_amdgcn_wmma_f32_16x16x32_bf16(
              false, afrag[mt], false, bfrag[nt], (short)0, acc[mt][nt],
              false, false);
    }

    // Next tile fully in LDS + everyone done reading current tile.
    asm volatile("s_wait_asynccnt 0x0" ::: "memory");
    __syncthreads();
    buf ^= 1;
  }

  // Epilogue. C/D layout: VGPR i -> M = i (lanes 0-15) / M = 8+i (lanes 16-31).
  const int mbase = bm + waveM * 64 + ((lane >> 4) << 3);
  const int nbase = bn + waveN * 64 + (lane & 15);
#pragma unroll
  for (int mt = 0; mt < 4; ++mt) {
#pragma unroll
    for (int nt = 0; nt < 4; ++nt) {
      const int cc = nbase + nt * 16;
      const float bv = bias[cc];
#pragma unroll
      for (int i = 0; i < 8; ++i) {
        const size_t idx = (size_t)(mbase + mt * 16 + i) * N + cc;
        float v = acc[mt][nt][i] + bv;
        if (EPILOGUE == 0) {
          v = fmaxf(v, 0.0f);
          v = v * v;
          ((__bf16*)Out)[idx] = (__bf16)v;
        } else {
          ((float*)Out)[idx] = v;
        }
      }
    }
  }
}

// ---------------------------------------------------------------------------
extern "C" void kernel_launch(void* const* d_in, const int* in_sizes, int n_in,
                              void* d_out, int out_size, void* d_ws,
                              size_t ws_size, hipStream_t stream) {
  const float* x  = (const float*)d_in[0];
  const float* w1 = (const float*)d_in[1];
  const float* b1 = (const float*)d_in[2];
  const float* w2 = (const float*)d_in[3];
  const float* b2 = (const float*)d_in[4];
  float* out = (float*)d_out;

  // Workspace layout (bf16): Xb | W1b | W2b | Ab
  __bf16* Xb  = (__bf16*)d_ws;
  __bf16* W1b = Xb  + (size_t)M_ROWS * D_DIM;
  __bf16* W2b = W1b + (size_t)H_DIM * D_DIM;
  __bf16* Ab  = W2b + (size_t)D_DIM * H_DIM;   // [M, H] activations

  // --- Phase 1: down-convert fp32 -> bf16 ---
  {
    const int nx = M_ROWS * D_DIM / 4;   // 4,194,304
    const int nw = H_DIM * D_DIM / 4;    // 4,194,304
    cvt_f32_to_bf16<<<nx / 256, 256, 0, stream>>>(x, Xb, nx);
    cvt_f32_to_bf16<<<nw / 256, 256, 0, stream>>>(w1, W1b, nw);
    cvt_f32_to_bf16<<<nw / 256, 256, 0, stream>>>(w2, W2b, nw);
  }

  // --- Phase 2: fc1 + bias + squared ReLU -> bf16 activations ---
  {
    dim3 grid(H_DIM / BN, M_ROWS / BM);  // (32, 64)
    gemm_wmma_bf16<0><<<grid, 256, 0, stream>>>(
        Xb, W1b, b1, (void*)Ab, M_ROWS, H_DIM, D_DIM);
  }

  // --- Phase 3: fc2 + bias -> fp32 output ---
  {
    dim3 grid(D_DIM / BN, M_ROWS / BM);  // (8, 64)
    gemm_wmma_bf16<1><<<grid, 256, 0, stream>>>(
        Ab, W2b, b2, (void*)out, M_ROWS, D_DIM, H_DIM);
  }
}
